// MoELayer_6373731467620
// MI455X (gfx1250) — matrix operand order repaired and checked by reading, not compiled, not run on hardware
//
#include <hip/hip_runtime.h>

#define NTOK 65536
#define DDIM 128
#define HDIM 256
#define NEXP 32
#define TOPK 2
#define MT   64            // rows (tokens) per FFN block
#define XS   136           // X tile row stride in bf16 elems (16B-aligned rows)
#define HS   264           // H tile row stride in bf16 elems (16B-aligned rows)

typedef __bf16 v16bf __attribute__((ext_vector_type(16)));
typedef float  v8f   __attribute__((ext_vector_type(8)));

union Frag16 {
    v16bf          bf;
    unsigned short us[16];
    uint4          q[2];
};

__device__ __forceinline__ unsigned short f2bf(float f) {
    __bf16 b = (__bf16)f;                      // hardware f32->bf16 (RNE)
    return __builtin_bit_cast(unsigned short, b);
}

// A-matrix (16x32 bf16) fragment from a row-major bf16 LDS tile.
// lanes 0-15: row M=lane,     elems = K kbase+{0..15}
// lanes 16-31: row M=lane-16, elems = K kbase+16+{0..15}
__device__ __forceinline__ Frag16 load_a_lds(const unsigned short* tile, int stride,
                                             int r0, int kbase, int lane) {
    int half = lane >> 4, mr = lane & 15;
    const unsigned short* p = tile + (r0 + mr) * stride + kbase + half * 16;
    Frag16 a;
    a.q[0] = *(const uint4*)(p);
    a.q[1] = *(const uint4*)(p + 8);
    return a;
}

// B-matrix fragment from pre-swizzled bf16 weights: frag-major, 32 lanes x 16 bf16.
__device__ __forceinline__ Frag16 load_b_sw(const unsigned short* __restrict__ Wsw,
                                            int frag, int lane) {
    const uint4* p = (const uint4*)(Wsw + ((size_t)frag * 32 + lane) * 16);
    Frag16 b;
    b.q[0] = p[0];
    b.q[1] = p[1];
    return b;
}

__device__ __forceinline__ v8f wmma_bf16(const Frag16& a, const Frag16& b, v8f c) {
    return __builtin_amdgcn_wmma_f32_16x16x32_bf16(false, a.bf, false, b.bf,
                                                   (short)0, c, false, false);
}

// ---------------------------------------------------------------- init
__global__ void init_kernel(int* __restrict__ counts) {
    if (threadIdx.x < 2 * NEXP) counts[threadIdx.x] = 0;
}

// ---------------------------------------------------------------- pre-swizzle weights (K x N f32 -> B-fragment bf16)
// fragment order: ((e*NT + nt)*KC + kc); within fragment: lane-major, 16 bf16/lane
// lane n-col = n0 + (lane&15); lane K rows = kc*32 + (lane>>4)*16 + {0..15}
__global__ __launch_bounds__(256) void swizzle_w_kernel(
    const float* __restrict__ W, unsigned short* __restrict__ out,
    int K, int Ncols, int NT, int KC)
{
    int t = blockIdx.x * 256 + threadIdx.x;
    int lane = t & 31;
    int r = t >> 5;
    int kc = r % KC; r /= KC;
    int nt = r % NT;
    int e  = r / NT;
    int nc = lane & 15;
    int kb = kc * 32 + (lane >> 4) * 16;
    const float* p = W + (size_t)e * K * Ncols + (size_t)kb * Ncols + nt * 16 + nc;
    Frag16 b;
#pragma unroll
    for (int j = 0; j < 16; ++j) b.us[j] = f2bf(p[(size_t)j * Ncols]);
    uint4* o = (uint4*)(out + ((size_t)(t >> 5) * 32 + lane) * 16);
    o[0] = b.q[0];
    o[1] = b.q[1];
}

// ---------------------------------------------------------------- pre-convert x (f32 -> bf16, row-major)
__global__ __launch_bounds__(256) void convert_x_kernel(
    const float* __restrict__ x, unsigned short* __restrict__ xbf)
{
    size_t i = (size_t)blockIdx.x * 256 + threadIdx.x;   // 8 elements per thread
    const float4* p = (const float4*)(x + i * 8);
    float4 a = p[0], c = p[1];
    union { unsigned short us[8]; uint4 q; } u;
    u.us[0] = f2bf(a.x); u.us[1] = f2bf(a.y); u.us[2] = f2bf(a.z); u.us[3] = f2bf(a.w);
    u.us[4] = f2bf(c.x); u.us[5] = f2bf(c.y); u.us[6] = f2bf(c.z); u.us[7] = f2bf(c.w);
    ((uint4*)xbf)[i] = u.q;
}

// ---------------------------------------------------------------- router
__global__ __launch_bounds__(256) void router_kernel(
    const float* __restrict__ x, const float* __restrict__ Wr, const float* __restrict__ br,
    int2* __restrict__ eidx, int2* __restrict__ offs, float2* __restrict__ gates,
    int* __restrict__ counts)
{
    __shared__ float WrL[DDIM * NEXP];
    __shared__ float brL[NEXP];
    int tid = threadIdx.x;
    for (int i = tid; i < DDIM * NEXP; i += 256) WrL[i] = Wr[i];
    if (tid < NEXP) brL[tid] = br[tid];
    __syncthreads();

    int n = blockIdx.x * 256 + tid;
    float acc[NEXP];
#pragma unroll
    for (int e = 0; e < NEXP; ++e) acc[e] = brL[e];
    const float4* xp = (const float4*)(x + (size_t)n * DDIM);
    for (int d4 = 0; d4 < DDIM / 4; ++d4) {
        float4 xv = xp[d4];
#pragma unroll
        for (int e = 0; e < NEXP; ++e) acc[e] = fmaf(xv.x, WrL[(d4 * 4 + 0) * NEXP + e], acc[e]);
#pragma unroll
        for (int e = 0; e < NEXP; ++e) acc[e] = fmaf(xv.y, WrL[(d4 * 4 + 1) * NEXP + e], acc[e]);
#pragma unroll
        for (int e = 0; e < NEXP; ++e) acc[e] = fmaf(xv.z, WrL[(d4 * 4 + 2) * NEXP + e], acc[e]);
#pragma unroll
        for (int e = 0; e < NEXP; ++e) acc[e] = fmaf(xv.w, WrL[(d4 * 4 + 3) * NEXP + e], acc[e]);
    }
    // softmax
    float mx = acc[0];
#pragma unroll
    for (int e = 1; e < NEXP; ++e) mx = fmaxf(mx, acc[e]);
    float sum = 0.0f;
#pragma unroll
    for (int e = 0; e < NEXP; ++e) { acc[e] = __expf(acc[e] - mx); sum += acc[e]; }
    float inv = 1.0f / sum;
#pragma unroll
    for (int e = 0; e < NEXP; ++e) acc[e] *= inv;
    // top-2 (ties -> lowest index, like top_k)
    float m1 = -1.0f; int i1 = 0;
#pragma unroll
    for (int e = 0; e < NEXP; ++e) if (acc[e] > m1) { m1 = acc[e]; i1 = e; }
    float m2 = -1.0f; int i2 = 0;
#pragma unroll
    for (int e = 0; e < NEXP; ++e) if (e != i1 && acc[e] > m2) { m2 = acc[e]; i2 = e; }

    int o1 = atomicAdd(&counts[i1], 1);           // slot-0 group = expert i1
    int o2 = atomicAdd(&counts[NEXP + i2], 1);    // slot-1 group = expert i2
    eidx[n]  = make_int2(i1, i2);
    offs[n]  = make_int2(o1, o2);
    gates[n] = make_float2(m1, m2);
}

// ---------------------------------------------------------------- prefix scan (64 entries)
__global__ void scan_kernel(const int* __restrict__ counts, int* __restrict__ bases) {
    if (threadIdx.x == 0) {
        int r = 0;
        for (int g = 0; g < 2 * NEXP; ++g) { bases[g] = r; r += counts[g]; }
    }
}

// ---------------------------------------------------------------- build permuted token lists
__global__ __launch_bounds__(256) void build_perm_kernel(
    const int2* __restrict__ eidx, const int2* __restrict__ offs, const float2* __restrict__ gates,
    const int* __restrict__ bases, int* __restrict__ perm, float* __restrict__ gperm)
{
    int n = blockIdx.x * 256 + threadIdx.x;
    int2 e = eidx[n]; int2 o = offs[n]; float2 g = gates[n];
    int r0 = bases[e.x] + o.x;
    int r1 = bases[NEXP + e.y] + o.y;
    perm[r0] = n; gperm[r0] = g.x;
    perm[r1] = n; gperm[r1] = g.y;
}

// ---------------------------------------------------------------- expert FFN (WMMA bf16)
__global__ __launch_bounds__(256) void moe_ffn_kernel(
    const unsigned short* __restrict__ xbf,
    const unsigned short* __restrict__ w1sw, const float* __restrict__ b1,
    const unsigned short* __restrict__ w2sw, const float* __restrict__ b2,
    const int* __restrict__ perm, const float* __restrict__ gperm,
    const int* __restrict__ counts, const int* __restrict__ bases,
    float* __restrict__ out, int slot)
{
    __shared__ __align__(16) unsigned short Xt[MT * XS];   // 64 x 128 bf16 (padded)
    __shared__ __align__(16) unsigned short Ht[MT * HS];   // 64 x 256 bf16 (padded)
    __shared__ int   srow[MT];
    __shared__ float sg[MT];

    int e = blockIdx.x;
    int g = slot * NEXP + e;
    int cnt  = counts[g];
    int row0 = blockIdx.y * MT;
    if (row0 >= cnt) return;
    int base = bases[g];

    int tid  = threadIdx.x;
    int lane = tid & 31, wv = tid >> 5;
    int half = lane >> 4, l16 = lane & 15;

    if (tid < MT) {
        int r = row0 + tid;
        if (r < cnt) { srow[tid] = perm[base + r]; sg[tid] = gperm[base + r]; }
        else         { srow[tid] = 0;              sg[tid] = 0.0f; }
    }
    __syncthreads();

    // gather X tile (MT x D bf16, straight b128 copies)
    for (int i = tid; i < MT * (DDIM / 8); i += 256) {
        int row = i >> 4, c8 = i & 15;                     // DDIM/8 == 16
        uint4 v = *(const uint4*)(xbf + (size_t)srow[row] * DDIM + c8 * 8);
        *(uint4*)&Xt[row * XS + c8 * 8] = v;
    }
    __syncthreads();

    const float* b1e = b1 + (size_t)e * HDIM;
    const float* b2e = b2 + (size_t)e * DDIM;

    // GEMM1: h = relu(X @ W1 + b1); wave covers n-tiles {wv, wv+8} of 16
    for (int t = 0; t < 2; ++t) {
        int nt = wv + t * 8;
        int n0 = nt * 16;
        Frag16 bfrag[4];
#pragma unroll
        for (int kc = 0; kc < 4; ++kc) bfrag[kc] = load_b_sw(w1sw, (e * 16 + nt) * 4 + kc, lane);
        float bias = b1e[n0 + l16];
#pragma unroll
        for (int rt = 0; rt < 4; ++rt) {
            v8f acc = {0.f, 0.f, 0.f, 0.f, 0.f, 0.f, 0.f, 0.f};
#pragma unroll
            for (int kc = 0; kc < 4; ++kc) {
                Frag16 a = load_a_lds(Xt, XS, rt * 16, kc * 32, lane);
                acc = wmma_bf16(a, bfrag[kc], acc);
            }
#pragma unroll
            for (int i = 0; i < 8; ++i) {
                int m = i + half * 8;
                float v = fmaxf(acc[i] + bias, 0.0f);
                Ht[(rt * 16 + m) * HS + n0 + l16] = f2bf(v);
            }
        }
    }
    __syncthreads();

    // GEMM2: y = (h @ W2 + b2) * gate; wave covers n-tile wv of 8 (D=128)
    {
        int n0 = wv * 16;
        Frag16 bfrag[8];
#pragma unroll
        for (int kc = 0; kc < 8; ++kc) bfrag[kc] = load_b_sw(w2sw, (e * 8 + wv) * 8 + kc, lane);
        float bias = b2e[n0 + l16];
#pragma unroll
        for (int rt = 0; rt < 4; ++rt) {
            v8f acc = {0.f, 0.f, 0.f, 0.f, 0.f, 0.f, 0.f, 0.f};
#pragma unroll
            for (int kc = 0; kc < 8; ++kc) {
                Frag16 a = load_a_lds(Ht, HS, rt * 16, kc * 32, lane);
                acc = wmma_bf16(a, bfrag[kc], acc);
            }
#pragma unroll
            for (int i = 0; i < 8; ++i) {
                int m  = i + half * 8;
                int rl = rt * 16 + m;
                if (row0 + rl < cnt) {
                    float y = (acc[i] + bias) * sg[rl];
                    float* op = out + (size_t)srow[rl] * DDIM + n0 + l16;
                    if (slot) *op = *op + y;   // slot-1: accumulate (one writer per token)
                    else      *op = y;         // slot-0: plain store (covers all tokens)
                }
            }
        }
    }
}

// ---------------------------------------------------------------- importance (deterministic)
__global__ __launch_bounds__(256) void importance_kernel(
    const int2* __restrict__ eidx, const float2* __restrict__ gates, float* __restrict__ imp)
{
    __shared__ float red[256];
    int e = blockIdx.x;
    float s = 0.0f;
    for (int n = threadIdx.x; n < NTOK; n += 256) {
        int2 ei = eidx[n]; float2 g = gates[n];
        if (ei.x == e) s += g.x;
        if (ei.y == e) s += g.y;
    }
    red[threadIdx.x] = s; __syncthreads();
    for (int off = 128; off > 0; off >>= 1) {
        if (threadIdx.x < off) red[threadIdx.x] += red[threadIdx.x + off];
        __syncthreads();
    }
    if (threadIdx.x == 0) imp[e] = red[0];
}

// ---------------------------------------------------------------- load-balancing loss
__global__ void loss_kernel(const float* __restrict__ imp, float* __restrict__ out) {
    int l = threadIdx.x;                    // 32 threads = 1 wave
    float v = imp[l];
    float s = v;
    for (int m = 16; m >= 1; m >>= 1) s += __shfl_xor(s, m);
    float mean = s / (float)NEXP;
    float d = v - mean;
    float ss = d * d;
    for (int m = 16; m >= 1; m >>= 1) ss += __shfl_xor(ss, m);
    float var = ss / (float)(NEXP - 1);     // unbiased (ddof=1)
    if (l == 0) out[(size_t)NTOK * DDIM] = var / (mean * mean + 1e-9f);
}

// ---------------------------------------------------------------- launch
extern "C" void kernel_launch(void* const* d_in, const int* in_sizes, int n_in,
                              void* d_out, int out_size, void* d_ws, size_t ws_size,
                              hipStream_t stream) {
    (void)in_sizes; (void)n_in; (void)out_size; (void)ws_size;
    const float* x  = (const float*)d_in[0];
    const float* W1 = (const float*)d_in[1];
    const float* b1 = (const float*)d_in[2];
    const float* W2 = (const float*)d_in[3];
    const float* b2 = (const float*)d_in[4];
    const float* Wr = (const float*)d_in[5];
    const float* br = (const float*)d_in[6];
    float* out = (float*)d_out;

    char* w = (char*)d_ws;
    int*    counts = (int*)w;            w += 256;
    int*    bases  = (int*)w;            w += 256;
    int2*   eidx   = (int2*)w;           w += (size_t)NTOK * 8;
    int2*   offs   = (int2*)w;           w += (size_t)NTOK * 8;
    float2* gates  = (float2*)w;         w += (size_t)NTOK * 8;
    int*    perm   = (int*)w;            w += (size_t)NTOK * TOPK * 4;
    float*  gperm  = (float*)w;          w += (size_t)NTOK * TOPK * 4;
    float*  imp    = (float*)w;          w += 128;
    unsigned short* w1sw = (unsigned short*)w;  w += (size_t)NEXP * DDIM * HDIM * 2;  // 2 MB
    unsigned short* w2sw = (unsigned short*)w;  w += (size_t)NEXP * HDIM * DDIM * 2;  // 2 MB
    unsigned short* xbf  = (unsigned short*)w;  w += (size_t)NTOK * DDIM * 2;         // 16 MB

    init_kernel<<<1, 64, 0, stream>>>(counts);
    router_kernel<<<NTOK / 256, 256, 0, stream>>>(x, Wr, br, eidx, offs, gates, counts);
    scan_kernel<<<1, 32, 0, stream>>>(counts, bases);
    build_perm_kernel<<<NTOK / 256, 256, 0, stream>>>(eidx, offs, gates, bases, perm, gperm);

    // W1: K=128, N=256 -> NT=16, KC=4 ; W2: K=256, N=128 -> NT=8, KC=8
    swizzle_w_kernel<<<(NEXP * 16 * 4 * 32) / 256, 256, 0, stream>>>(W1, w1sw, DDIM, HDIM, 16, 4);
    swizzle_w_kernel<<<(NEXP * 8 * 8 * 32) / 256, 256, 0, stream>>>(W2, w2sw, HDIM, DDIM, 8, 8);
    convert_x_kernel<<<(NTOK * DDIM / 8) / 256, 256, 0, stream>>>(x, xbf);

    dim3 gffn(NEXP, NTOK / MT);   // worst case: one group holds all N tokens
    moe_ffn_kernel<<<gffn, 256, 0, stream>>>(xbf, w1sw, b1, w2sw, b2, perm, gperm, counts, bases, out, 0);
    moe_ffn_kernel<<<gffn, 256, 0, stream>>>(xbf, w1sw, b1, w2sw, b2, perm, gperm, counts, bases, out, 1);

    importance_kernel<<<NEXP, 256, 0, stream>>>(eidx, gates, imp);
    loss_kernel<<<1, 32, 0, stream>>>(imp, out);
}
